// BaselineGNN_34565896798973
// MI455X (gfx1250) — matrix operand order repaired and checked by reading, not compiled, not run on hardware
//
#include <hip/hip_runtime.h>

#define N_NODES  50000
#define N_EDGES  800000
#define N_GRAPHS 1000
#define IN_CH    78
#define HID      128
#define HID2     64
#define BN_EPS   1e-5f

typedef __attribute__((ext_vector_type(16))) __bf16 v16bf;
typedef __attribute__((ext_vector_type(8)))  float  v8f;

// ---------------------------------------------------------------- utilities
__global__ void zero_f32(float* __restrict__ p, size_t n) {
    size_t i = (size_t)blockIdx.x * blockDim.x + threadIdx.x;
    if (i < n) p[i] = 0.0f;
}

// pad x: [N,78] fp32 -> [N,128] fp32 + bf16 (cols 78..127 zero)
__global__ void pad_x(const float* __restrict__ x, float* __restrict__ hF,
                      __bf16* __restrict__ hB) {
    int i = blockIdx.x, c = threadIdx.x;          // grid N_NODES, block HID
    float v = (c < IN_CH) ? x[(size_t)i * IN_CH + c] : 0.0f;
    hF[(size_t)i * HID + c] = v;
    hB[(size_t)i * HID + c] = (__bf16)v;
}

// transpose+pad weight (Kin,HID) fp32 -> (HID,128) bf16, WT[n*128+k]
__global__ void pack_w(const float* __restrict__ W, __bf16* __restrict__ WT, int Kin) {
    int n = blockIdx.x, k = threadIdx.x;          // grid HID, block HID
    WT[(size_t)n * HID + k] = (k < Kin) ? (__bf16)W[(size_t)k * HID + n] : (__bf16)0.0f;
}

// ---------------------------------------------------------------- graph side
__global__ void degree_k(const long long* __restrict__ ei, float* __restrict__ deg) {
    int e = blockIdx.x * blockDim.x + threadIdx.x;
    if (e < N_EDGES) atomicAdd(&deg[(int)ei[(size_t)N_EDGES + e]], 1.0f);
}

// one wave (32 lanes) per edge; float4 per lane covers 128 channels
__global__ void edge_aggr(const float* __restrict__ hF, const long long* __restrict__ ei,
                          float* __restrict__ agg) {
    int t = blockIdx.x * blockDim.x + threadIdx.x;
    int e = t >> 5, lane = t & 31;
    if (e >= N_EDGES) return;
    int s = (int)ei[e];
    int d = (int)ei[(size_t)N_EDGES + e];
    float4 v = ((const float4*)(hF + (size_t)s * HID))[lane];
    float* dp = agg + (size_t)d * HID + lane * 4;
    atomicAdd(dp + 0, v.x);
    atomicAdd(dp + 1, v.y);
    atomicAdd(dp + 2, v.z);
    atomicAdd(dp + 3, v.w);
}

// agg / max(deg,1) -> bf16
__global__ void agg_norm(const float* __restrict__ agg, const float* __restrict__ deg,
                         __bf16* __restrict__ aggB) {
    int i = blockIdx.x, c = threadIdx.x;          // grid N_NODES, block HID
    float inv = 1.0f / fmaxf(deg[i], 1.0f);
    aggB[(size_t)i * HID + c] = (__bf16)(agg[(size_t)i * HID + c] * inv);
}

// ---------------------------------------------------------------- WMMA GEMM
// lin = aggB @ WnT^T + hB @ WrT^T + bias   (all K=128, bf16 in / fp32 acc)
// one 16x16 tile per wave; 8 waves/block cover the full 128 output columns.
union U32x16 { v16bf v; uint4 u[2]; };

__global__ __launch_bounds__(256)
void gemm_sage(const __bf16* __restrict__ aggB, const __bf16* __restrict__ hB,
               const __bf16* __restrict__ WnT,  const __bf16* __restrict__ WrT,
               const float* __restrict__ bias,  float* __restrict__ lin) {
    int lane = threadIdx.x & 31;
    int tn   = threadIdx.x >> 5;                  // N-tile 0..7
    int tm   = blockIdx.x;                        // M-tile 0..3124
    int half = lane >> 4;
    int l15  = lane & 15;
    int am   = tm * 16 + l15;                     // A row this lane feeds
    int bn   = tn * 16 + l15;                     // B column (row of WT)

    v8f c = {};
    const __bf16* a0 = aggB + (size_t)am * HID;
    const __bf16* a1 = hB   + (size_t)am * HID;
    const __bf16* w0 = WnT  + (size_t)bn * HID;
    const __bf16* w1 = WrT  + (size_t)bn * HID;

    #pragma unroll
    for (int kk = 0; kk < HID; kk += 32) {
        // A 16x32 bf16: lane half selects K{0..7,16..23} vs K{8..15,24..31}
        U32x16 a, b;
        const uint4* ap = (const uint4*)(a0 + kk + half * 8);
        a.u[0] = ap[0]; a.u[1] = ap[2];
        // B 32x16 bf16 (pre-transposed): lane half selects K{0..7,16..23} vs K{8..15,24..31}
        const uint4* bp = (const uint4*)(w0 + kk + half * 8);
        b.u[0] = bp[0]; b.u[1] = bp[2];
        c = __builtin_amdgcn_wmma_f32_16x16x32_bf16(false, a.v, false, b.v,
                                                    (short)0, c, false, false);

        ap = (const uint4*)(a1 + kk + half * 8);
        a.u[0] = ap[0]; a.u[1] = ap[2];
        bp = (const uint4*)(w1 + kk + half * 8);
        b.u[0] = bp[0]; b.u[1] = bp[2];
        c = __builtin_amdgcn_wmma_f32_16x16x32_bf16(false, a.v, false, b.v,
                                                    (short)0, c, false, false);
    }

    int col = tn * 16 + l15;
    float bb = bias[col];
    #pragma unroll
    for (int r = 0; r < 8; ++r) {                 // D: lane=N, vgpr=M (+8 for hi half)
        int row = tm * 16 + r + 8 * half;
        lin[(size_t)row * HID + col] = c[r] + bb;
    }
}

// ---------------------------------------------------------------- batch norm
__global__ void bn_stats(const float* __restrict__ lin, float* __restrict__ mu,
                         float* __restrict__ rinv) {
    int ch = blockIdx.x;                          // grid HID, block 256
    __shared__ float s1[256], s2[256];
    float a = 0.0f, b = 0.0f;
    for (int i = threadIdx.x; i < N_NODES; i += 256) {
        float v = lin[(size_t)i * HID + ch];
        a += v; b += v * v;
    }
    s1[threadIdx.x] = a; s2[threadIdx.x] = b;
    __syncthreads();
    for (int s = 128; s > 0; s >>= 1) {
        if (threadIdx.x < s) { s1[threadIdx.x] += s1[threadIdx.x + s];
                               s2[threadIdx.x] += s2[threadIdx.x + s]; }
        __syncthreads();
    }
    if (threadIdx.x == 0) {
        float m   = s1[0] / (float)N_NODES;
        float var = s2[0] / (float)N_NODES - m * m;
        mu[ch]   = m;
        rinv[ch] = rsqrtf(var + BN_EPS);
    }
}

__global__ void bn_relu(const float* __restrict__ lin, const float* __restrict__ mu,
                        const float* __restrict__ rinv, const float* __restrict__ g,
                        const float* __restrict__ be, float* __restrict__ hF,
                        __bf16* __restrict__ hB) {
    int i = blockIdx.x, c = threadIdx.x;          // grid N_NODES, block HID
    float y = (lin[(size_t)i * HID + c] - mu[c]) * rinv[c] * g[c] + be[c];
    y = fmaxf(y, 0.0f);
    hF[(size_t)i * HID + c] = y;
    hB[(size_t)i * HID + c] = (__bf16)y;
}

// ---------------------------------------------------------------- pool + head
__global__ void pool_accum(const float* __restrict__ hF, const long long* __restrict__ batch,
                           float* __restrict__ pooled, float* __restrict__ cnt) {
    int i = blockIdx.x, c = threadIdx.x;          // grid N_NODES, block HID
    int g = (int)batch[i];
    atomicAdd(&pooled[(size_t)g * HID + c], hF[(size_t)i * HID + c]);
    if (c == 0) atomicAdd(&cnt[g], 1.0f);
}

__global__ void head_k(const float* __restrict__ pooled, const float* __restrict__ cnt,
                       const float* __restrict__ Wh1, const float* __restrict__ bh1,
                       const float* __restrict__ Wh2, const float* __restrict__ bh2,
                       float* __restrict__ out) {
    int g = blockIdx.x, t = threadIdx.x;          // grid N_GRAPHS, block HID2
    float invc = 1.0f / fmaxf(cnt[g], 1.0f);
    float acc = bh1[t];
    for (int c = 0; c < HID; ++c)
        acc += pooled[(size_t)g * HID + c] * invc * Wh1[c * HID2 + t];
    acc = fmaxf(acc, 0.0f);
    __shared__ float sh[HID2];
    sh[t] = acc * Wh2[t];
    __syncthreads();
    for (int s = HID2 / 2; s > 0; s >>= 1) {
        if (t < s) sh[t] += sh[t + s];
        __syncthreads();
    }
    if (t == 0) out[g] = sh[0] + bh2[0];
}

// ---------------------------------------------------------------- launcher
extern "C" void kernel_launch(void* const* d_in, const int* in_sizes, int n_in,
                              void* d_out, int out_size, void* d_ws, size_t ws_size,
                              hipStream_t stream) {
    (void)in_sizes; (void)n_in; (void)out_size; (void)ws_size;

    const float*     x  = (const float*)d_in[0];
    const long long* ei = (const long long*)d_in[1];   // reference dtype int64
    const long long* bt = (const long long*)d_in[2];
    const float* Wh1 = (const float*)d_in[18];
    const float* bh1 = (const float*)d_in[19];
    const float* Wh2 = (const float*)d_in[20];
    const float* bh2 = (const float*)d_in[21];
    float* out = (float*)d_out;

    // workspace carve-up (256B aligned)
    char* w = (char*)d_ws;
    auto alloc = [&](size_t bytes) { char* p = w; w += (bytes + 255) & ~(size_t)255; return p; };
    float*  hF     = (float*) alloc((size_t)N_NODES * HID * 4);
    __bf16* hB     = (__bf16*)alloc((size_t)N_NODES * HID * 2);
    float*  agg    = (float*) alloc((size_t)N_NODES * HID * 4);
    __bf16* aggB   = (__bf16*)alloc((size_t)N_NODES * HID * 2);
    float*  deg    = (float*) alloc((size_t)N_NODES * 4);
    __bf16* WnT    = (__bf16*)alloc((size_t)HID * HID * 2);
    __bf16* WrT    = (__bf16*)alloc((size_t)HID * HID * 2);
    float*  mu     = (float*) alloc(HID * 4);
    float*  rinv   = (float*) alloc(HID * 4);
    float*  pooled = (float*) alloc((size_t)N_GRAPHS * HID * 4);
    float*  cnt    = (float*) alloc((size_t)N_GRAPHS * 4);
    float*  lin    = agg;   // alias: agg fp32 dead once aggB is built

    const size_t nFeat = (size_t)N_NODES * HID;

    // input padding + one-time degrees
    pad_x<<<N_NODES, HID, 0, stream>>>(x, hF, hB);
    zero_f32<<<(N_NODES + 255) / 256, 256, 0, stream>>>(deg, N_NODES);
    degree_k<<<(N_EDGES + 255) / 256, 256, 0, stream>>>(ei, deg);

    for (int l = 0; l < 3; ++l) {
        const float* W_n  = (const float*)d_in[3 + 5 * l];
        const float* bias = (const float*)d_in[4 + 5 * l];
        const float* W_r  = (const float*)d_in[5 + 5 * l];
        const float* gam  = (const float*)d_in[6 + 5 * l];
        const float* bet  = (const float*)d_in[7 + 5 * l];
        int Kin = (l == 0) ? IN_CH : HID;

        pack_w<<<HID, HID, 0, stream>>>(W_n, WnT, Kin);
        pack_w<<<HID, HID, 0, stream>>>(W_r, WrT, Kin);

        zero_f32<<<(int)((nFeat + 255) / 256), 256, 0, stream>>>(agg, nFeat);
        edge_aggr<<<(N_EDGES * 32 + 255) / 256, 256, 0, stream>>>(hF, ei, agg);
        agg_norm<<<N_NODES, HID, 0, stream>>>(agg, deg, aggB);

        gemm_sage<<<N_NODES / 16, 256, 0, stream>>>(aggB, hB, WnT, WrT, bias, lin);

        bn_stats<<<HID, 256, 0, stream>>>(lin, mu, rinv);
        bn_relu<<<N_NODES, HID, 0, stream>>>(lin, mu, rinv, gam, bet, hF, hB);
    }

    zero_f32<<<(N_GRAPHS * HID + 255) / 256, 256, 0, stream>>>(pooled, (size_t)N_GRAPHS * HID);
    zero_f32<<<(N_GRAPHS + 255) / 256, 256, 0, stream>>>(cnt, N_GRAPHS);
    pool_accum<<<N_NODES, HID, 0, stream>>>(hF, bt, pooled, cnt);
    head_k<<<N_GRAPHS, HID2, 0, stream>>>(pooled, cnt, Wh1, bh1, Wh2, bh2, out);
}
